// GraphClassifier_20375324852532
// MI455X (gfx1250) — matrix operand order repaired and checked by reading, not compiled, not run on hardware
//
#include <hip/hip_runtime.h>
#include <hip/hip_bf16.h>
#include <float.h>
#include <stdint.h>

typedef __bf16 bf16_t;
typedef __attribute__((ext_vector_type(16))) __bf16 v16bf;
typedef __attribute__((ext_vector_type(8)))  float  v8f;

#define N_EDGES     800000
#define N_NODES     50000
#define TOT_ROWS    (N_EDGES + N_NODES)     /* 850000, divisible by 16 */
#define EDGE_DIM    64
#define HIDDEN      128
#define NUM_CLASSES 10
#define NUM_GRAPHS  128

#define N_TILES     (TOT_ROWS / 16)         /* 53125 16-row tiles */
#define WAVES_PER_BLOCK 4
#define MLP_BLOCKS  2048
#define LDS_STRIDE  (HIDDEN + 4)            /* pad to dodge bank conflicts */

// ---------------------------------------------------------------------------
// Split an fp32 value into bf16 hi + bf16 lo (hi+lo ~= x to ~2^-16 rel).
// ---------------------------------------------------------------------------
__device__ __forceinline__ void bf16_split(float x, __bf16& hi, __bf16& lo) {
    __bf16 h = (__bf16)x;
    hi = h;
    lo = (__bf16)(x - (float)h);
}

// ---------------------------------------------------------------------------
// Init: node scratch = -FLT_MAX (scatter-max identity).
// ---------------------------------------------------------------------------
__global__ void init_kernel(float* __restrict__ node) {
    long i = (long)blockIdx.x * blockDim.x + threadIdx.x;
    long stride = (long)gridDim.x * blockDim.x;
    for (long t = i; t < (long)N_NODES * HIDDEN; t += stride)
        node[t] = -FLT_MAX;
}

// ---------------------------------------------------------------------------
// Prep: repack a KxN fp32 weight into WMMA B-fragment layout, bf16 hi/lo.
// B fragment layout (16x16x32 bf16): lane holds column n = nt*16 + (lane&15);
// half j (0..15) holds k = kt*32 + ((lane>>4)<<3) + (j&7) + ((j>>3)<<4).
// Stored as frag[(kt*NT+nt)*32 + lane][16] -> one contiguous 32B block/lane.
// ---------------------------------------------------------------------------
__global__ void prep_kernel(const float* __restrict__ W,
                            bf16_t* __restrict__ hi, bf16_t* __restrict__ lo,
                            int KT, int NT, int N) {
    int t = blockIdx.x * blockDim.x + threadIdx.x;     // one thread per (frag, lane)
    int total = KT * NT * 32;
    if (t >= total) return;
    int lane = t & 31;
    int f    = t >> 5;
    int nt   = f % NT;
    int kt   = f / NT;
    int n  = nt * 16 + (lane & 15);
    int kb = kt * 32 + ((lane >> 4) << 3);
    #pragma unroll
    for (int j = 0; j < 16; ++j) {
        int k = kb + (j & 7) + ((j >> 3) << 4);
        float x = W[k * N + n];
        __bf16 h, l;
        bf16_split(x, h, l);
        hi[t * 16 + j] = h;
        lo[t * 16 + j] = l;
    }
}

// ---------------------------------------------------------------------------
// Hidden layer (128->128) on a wave-private 16x128 LDS tile, bf16x3 WMMA.
// Reads all of hb first (4 K-steps), then overwrites hb with the result.
// Same-wave LDS ops are in-order, so no barrier is required.
// kt loop is intentionally NOT unrolled to bound VGPR live ranges.
// ---------------------------------------------------------------------------
template <bool RELU>
__device__ __forceinline__ void hidden_layer(float* hb,
                                             const float* __restrict__ bias,
                                             const bf16_t* __restrict__ whi,
                                             const bf16_t* __restrict__ wlo,
                                             int lane) {
    const int ncol = lane & 15;
    const int half = lane >> 4;
    const int mof  = half * 8;

    v8f acc[8];
    #pragma unroll
    for (int nt = 0; nt < 8; ++nt) {
        float b = bias[nt * 16 + ncol];
        #pragma unroll
        for (int i = 0; i < 8; ++i) acc[nt][i] = b;
    }

    #pragma unroll 1
    for (int kt = 0; kt < HIDDEN / 32; ++kt) {
        // A fragment from wave-private LDS tile (fp32 -> bf16 hi/lo).
        v16bf ahi, alo;
        const int m  = lane & 15;
        const int kb = kt * 32 + (half << 3);
        #pragma unroll
        for (int j = 0; j < 16; ++j) {
            int k = kb + (j & 7) + ((j >> 3) << 4);
            float x = hb[m * LDS_STRIDE + k];
            __bf16 h, l;
            bf16_split(x, h, l);
            ahi[j] = h;
            alo[j] = l;
        }
        #pragma unroll
        for (int nt = 0; nt < 8; ++nt) {
            const int fi = (kt * 8 + nt) * 32 + lane;
            v16bf bhi = ((const v16bf*)whi)[fi];
            v16bf blo = ((const v16bf*)wlo)[fi];
            acc[nt] = __builtin_amdgcn_wmma_f32_16x16x32_bf16(false, ahi, false, bhi, (short)0, acc[nt], false, false);
            acc[nt] = __builtin_amdgcn_wmma_f32_16x16x32_bf16(false, ahi, false, blo, (short)0, acc[nt], false, false);
            acc[nt] = __builtin_amdgcn_wmma_f32_16x16x32_bf16(false, alo, false, bhi, (short)0, acc[nt], false, false);
        }
        __builtin_amdgcn_sched_barrier(0);   // keep per-kt live sets separate
    }

    // Write result back (all reads of hb completed above).
    #pragma unroll
    for (int nt = 0; nt < 8; ++nt) {
        #pragma unroll
        for (int i = 0; i < 8; ++i) {
            float v = acc[nt][i];
            if (RELU) v = fmaxf(v, 0.0f);
            hb[(mof + i) * LDS_STRIDE + nt * 16 + ncol] = v;
        }
    }
    __builtin_amdgcn_sched_barrier(0);
}

// ---------------------------------------------------------------------------
// Fused edge MLP: 16-row tiles, 3 layers in registers/LDS, scatter-max out.
// ---------------------------------------------------------------------------
__global__ __launch_bounds__(WAVES_PER_BLOCK * 32)
void edge_mlp_kernel(const float* __restrict__ edge_attr,
                     const float* __restrict__ b1,
                     const float* __restrict__ b2,
                     const float* __restrict__ b3,
                     const bf16_t* __restrict__ w1hi, const bf16_t* __restrict__ w1lo,
                     const bf16_t* __restrict__ w2hi, const bf16_t* __restrict__ w2lo,
                     const bf16_t* __restrict__ w3hi, const bf16_t* __restrict__ w3lo,
                     const int* __restrict__ dst_idx,   /* edge_index row 1 */
                     float* __restrict__ node) {
    __shared__ float hbuf_all[WAVES_PER_BLOCK][16 * LDS_STRIDE];

    const int lane = threadIdx.x & 31;
    const int wave = threadIdx.x >> 5;
    float* hb = hbuf_all[wave];

    const int totalWaves = gridDim.x * WAVES_PER_BLOCK;
    const int waveGlobal = blockIdx.x * WAVES_PER_BLOCK + wave;
    const int iters = (N_TILES + totalWaves - 1) / totalWaves;

    const int ncol = lane & 15;
    const int half = lane >> 4;
    const int mof  = half * 8;

    #pragma unroll 1
    for (int it = 0; it < iters; ++it) {
        const int  tile    = waveGlobal + it * totalWaves;
        const long rowBase = (long)tile * 16;

        // ---------------- Layer 1: 64 -> 128 ----------------
        v8f acc[8];
        #pragma unroll
        for (int nt = 0; nt < 8; ++nt) {
            float b = b1[nt * 16 + ncol];
            #pragma unroll
            for (int i = 0; i < 8; ++i) acc[nt][i] = b;
        }
        #pragma unroll 1
        for (int kt = 0; kt < EDGE_DIM / 32; ++kt) {
            v16bf ahi, alo;
            {
                const long r   = rowBase + (lane & 15);
                const bool inE = (r < (long)N_EDGES);   // self-loop rows are zeros
                const float* rp = edge_attr + r * EDGE_DIM;
                const int kb = kt * 32 + (half << 3);
                #pragma unroll
                for (int j = 0; j < 16; ++j) {
                    int k = kb + (j & 7) + ((j >> 3) << 4);
                    float x = inE ? rp[k] : 0.0f;
                    __bf16 h, l;
                    bf16_split(x, h, l);
                    ahi[j] = h;
                    alo[j] = l;
                }
            }
            #pragma unroll
            for (int nt = 0; nt < 8; ++nt) {
                const int fi = (kt * 8 + nt) * 32 + lane;
                v16bf bhi = ((const v16bf*)w1hi)[fi];
                v16bf blo = ((const v16bf*)w1lo)[fi];
                acc[nt] = __builtin_amdgcn_wmma_f32_16x16x32_bf16(false, ahi, false, bhi, (short)0, acc[nt], false, false);
                acc[nt] = __builtin_amdgcn_wmma_f32_16x16x32_bf16(false, ahi, false, blo, (short)0, acc[nt], false, false);
                acc[nt] = __builtin_amdgcn_wmma_f32_16x16x32_bf16(false, alo, false, bhi, (short)0, acc[nt], false, false);
            }
            __builtin_amdgcn_sched_barrier(0);
        }
        #pragma unroll
        for (int nt = 0; nt < 8; ++nt) {
            #pragma unroll
            for (int i = 0; i < 8; ++i)
                hb[(mof + i) * LDS_STRIDE + nt * 16 + ncol] = fmaxf(acc[nt][i], 0.0f);
        }
        __builtin_amdgcn_sched_barrier(0);

        // ---------------- Layers 2 & 3: 128 -> 128 ----------------
        hidden_layer<true >(hb, b2, w2hi, w2lo, lane);
        hidden_layer<false>(hb, b3, w3hi, w3lo, lane);

        // ---------------- Scatter-max to destination nodes ----------------
        if (tile < N_TILES) {
            #pragma unroll 1
            for (int m = 0; m < 16; ++m) {
                const long r = rowBase + m;
                const long d = (r < (long)N_EDGES) ? (long)dst_idx[r]
                                                   : (r - (long)N_EDGES);
                float* nrow = node + d * HIDDEN;
                #pragma unroll
                for (int c = lane; c < HIDDEN; c += 32)
                    atomicMax(nrow + c, hb[m * LDS_STRIDE + c]);
            }
        }
        __builtin_amdgcn_sched_barrier(0);
    }
}

// ---------------------------------------------------------------------------
// Mean-pool (batch is sorted -> binary search per graph, deterministic sums)
// + 2-layer fp32 classifier. One 128-thread block per graph.
// ---------------------------------------------------------------------------
__global__ __launch_bounds__(HIDDEN)
void pool_cls_kernel(const float* __restrict__ node,
                     const int* __restrict__ batch,
                     const float* __restrict__ Wc1, const float* __restrict__ bc1,
                     const float* __restrict__ Wc2, const float* __restrict__ bc2,
                     float* __restrict__ out) {
    const int gi = blockIdx.x;
    const int c  = threadIdx.x;

    // lower_bound in sorted batch[]
    auto lower = [&](int key) {
        int lo = 0, hi = N_NODES;
        while (lo < hi) {
            int mid = (lo + hi) >> 1;
            if (batch[mid] < key) lo = mid + 1; else hi = mid;
        }
        return lo;
    };
    const int s = lower(gi);
    const int e = lower(gi + 1);

    float sum = 0.0f;
    for (int r = s; r < e; ++r)
        sum += node[(long)r * HIDDEN + c];
    const float cnt = (float)(e - s);

    __shared__ float grow[HIDDEN];
    __shared__ float t1[HIDDEN];
    grow[c] = sum / fmaxf(cnt, 1.0f);
    __syncthreads();

    float a = bc1[c];
    for (int k = 0; k < HIDDEN; ++k)
        a += grow[k] * Wc1[k * HIDDEN + c];
    t1[c] = fmaxf(a, 0.0f);
    __syncthreads();

    if (c < NUM_CLASSES) {
        float o = bc2[c];
        for (int k = 0; k < HIDDEN; ++k)
            o += t1[k] * Wc2[k * NUM_CLASSES + c];
        out[gi * NUM_CLASSES + c] = o;
    }
}

// ---------------------------------------------------------------------------
extern "C" void kernel_launch(void* const* d_in, const int* in_sizes, int n_in,
                              void* d_out, int out_size, void* d_ws, size_t ws_size,
                              hipStream_t stream) {
    (void)in_sizes; (void)n_in; (void)out_size; (void)ws_size;

    const float* edge_attr = (const float*)d_in[0];
    const float* W1  = (const float*)d_in[1];
    const float* b1  = (const float*)d_in[2];
    const float* W2  = (const float*)d_in[3];
    const float* b2  = (const float*)d_in[4];
    const float* W3  = (const float*)d_in[5];
    const float* b3  = (const float*)d_in[6];
    const float* Wc1 = (const float*)d_in[7];
    const float* bc1 = (const float*)d_in[8];
    const float* Wc2 = (const float*)d_in[9];
    const float* bc2 = (const float*)d_in[10];
    const int* edge_index = (const int*)d_in[11];   // [2, E]; row 1 = dst
    const int* batch      = (const int*)d_in[12];
    float* out = (float*)d_out;

    // Workspace carve-up (256B aligned).
    char* ws = (char*)d_ws;
    size_t off = 0;
    auto take = [&](size_t bytes) {
        size_t r = off;
        off += (bytes + 255) & ~(size_t)255;
        return r;
    };
    bf16_t* w1hi = (bf16_t*)(ws + take((size_t)EDGE_DIM * HIDDEN * sizeof(bf16_t)));
    bf16_t* w1lo = (bf16_t*)(ws + take((size_t)EDGE_DIM * HIDDEN * sizeof(bf16_t)));
    bf16_t* w2hi = (bf16_t*)(ws + take((size_t)HIDDEN * HIDDEN * sizeof(bf16_t)));
    bf16_t* w2lo = (bf16_t*)(ws + take((size_t)HIDDEN * HIDDEN * sizeof(bf16_t)));
    bf16_t* w3hi = (bf16_t*)(ws + take((size_t)HIDDEN * HIDDEN * sizeof(bf16_t)));
    bf16_t* w3lo = (bf16_t*)(ws + take((size_t)HIDDEN * HIDDEN * sizeof(bf16_t)));
    float*  node = (float*)(ws + take((size_t)N_NODES * HIDDEN * sizeof(float)));

    // 1) node <- -inf
    init_kernel<<<4096, 256, 0, stream>>>(node);

    // 2) repack weights into WMMA B-fragment layout (bf16 hi/lo)
    prep_kernel<<<(2 * 8 * 32 + 255) / 256, 256, 0, stream>>>(W1, w1hi, w1lo, 2, 8, HIDDEN);
    prep_kernel<<<(4 * 8 * 32 + 255) / 256, 256, 0, stream>>>(W2, w2hi, w2lo, 4, 8, HIDDEN);
    prep_kernel<<<(4 * 8 * 32 + 255) / 256, 256, 0, stream>>>(W3, w3hi, w3lo, 4, 8, HIDDEN);

    // 3) fused edge MLP + scatter-max
    edge_mlp_kernel<<<MLP_BLOCKS, WAVES_PER_BLOCK * 32, 0, stream>>>(
        edge_attr, b1, b2, b3,
        w1hi, w1lo, w2hi, w2lo, w3hi, w3lo,
        edge_index + N_EDGES, node);

    // 4) deterministic mean-pool + classifier
    pool_cls_kernel<<<NUM_GRAPHS, HIDDEN, 0, stream>>>(node, batch, Wc1, bc1, Wc2, bc2, out);
}
